// TtMambaSSM_9526237463158
// MI455X (gfx1250) — compile-verified, matrix operand order checked by reading, not run on hardware
//
#include <hip/hip_runtime.h>
#include <hip/hip_bf16.h>
#include <math.h>

typedef __attribute__((ext_vector_type(16))) _Float16 v16h;
typedef __attribute__((ext_vector_type(8)))  float    v8f;

#define B_SZ    256
#define D_INNER 5120
#define D_STATE 16
#define DT_RANK 160
#define NCAT    192   // DT_RANK + 2*D_STATE  (P | Bm | Cm)

// Convert 8 consecutive f32 -> 8 f16 lanes of a v16h starting at `off`.
__device__ __forceinline__ void cvt8(v16h& v, int off, const float* __restrict__ p) {
  float4 f0 = ((const float4*)p)[0];
  float4 f1 = ((const float4*)p)[1];
  v[off + 0] = (_Float16)f0.x; v[off + 1] = (_Float16)f0.y;
  v[off + 2] = (_Float16)f0.z; v[off + 3] = (_Float16)f0.w;
  v[off + 4] = (_Float16)f1.x; v[off + 5] = (_Float16)f1.y;
  v[off + 6] = (_Float16)f1.z; v[off + 7] = (_Float16)f1.w;
}

// ---------------------------------------------------------------------------
// Kernel 0: Aneg[d][n] = -exp(A_log[d][n])   (batch-invariant, hoisted)
// ---------------------------------------------------------------------------
__global__ __launch_bounds__(256) void aneg_kernel(
    const float* __restrict__ A_log, float* __restrict__ Aneg, int n) {
  int i = blockIdx.x * blockDim.x + threadIdx.x;
  if (i < n) Aneg[i] = -__expf(A_log[i]);
}

// ---------------------------------------------------------------------------
// Kernel 1: R[256,192] = x[256,5120] @ [W_x_dt | W_B | W_C]^T
// One 16x16 output tile per block; 4 waves split K=5120; LDS reduce.
// ---------------------------------------------------------------------------
__global__ __launch_bounds__(128) void xproj_wmma_kernel(
    const float* __restrict__ x,
    const float* __restrict__ W_x_dt,
    const float* __restrict__ W_B,
    const float* __restrict__ W_C,
    float* __restrict__ R) {
  const int lane  = threadIdx.x & 31;
  const int wave  = threadIdx.x >> 5;
  const int tileN = blockIdx.x;   // 0..11
  const int tileM = blockIdx.y;   // 0..15

  const int mn    = lane & 15;            // A row / B col within tile
  const int khalf = (lane >> 4) * 8;      // K sub-offset per lane half

  const int brow = tileM * 16 + mn;       // batch row
  const int ncol = tileN * 16 + mn;       // concatenated output feature

  const float* wrow;
  if (ncol < DT_RANK)                 wrow = W_x_dt + (size_t)ncol * D_INNER;
  else if (ncol < DT_RANK + D_STATE)  wrow = W_B + (size_t)(ncol - DT_RANK) * D_INNER;
  else                                wrow = W_C + (size_t)(ncol - DT_RANK - D_STATE) * D_INNER;

  const float* arow = x + (size_t)brow * D_INNER;

  v8f acc = {};
  const int KPW     = D_INNER / 4;        // 1280 K per wave
  const int k_begin = wave * KPW;
  const int k_end   = k_begin + KPW;
  for (int k0 = k_begin; k0 < k_end; k0 += 32) {
    v16h a = {}, b = {};
    cvt8(a, 0, arow + k0 + khalf);
    cvt8(a, 8, arow + k0 + khalf + 16);
    cvt8(b, 0, wrow + k0 + khalf);
    cvt8(b, 8, wrow + k0 + khalf + 16);
    acc = __builtin_amdgcn_wmma_f32_16x16x32_f16(
        false, a, false, b, (short)0, acc, false, false);
  }

  __shared__ float red[4][256];
  #pragma unroll
  for (int v = 0; v < 8; ++v) {
    int m = v + ((lane >> 4) << 3);
    red[wave][m * 16 + mn] = acc[v];
  }
  __syncthreads();
  for (int e = threadIdx.x; e < 256; e += 128) {
    float s = red[0][e] + red[1][e] + red[2][e] + red[3][e];
    int m = e >> 4, n = e & 15;
    R[(size_t)(tileM * 16 + m) * NCAT + (tileN * 16 + n)] = s;
  }
}

// ---------------------------------------------------------------------------
// Kernel 2: dt[256,5120] = softplus(P[256,160] @ W_dt^T + b_dt)
// P = first 160 columns of R. K=160 -> 5 unrolled WMMAs. One wave per tile.
// dt is written into d_out (same shape as y; overwritten later in-place).
// ---------------------------------------------------------------------------
__global__ __launch_bounds__(32) void dtproj_wmma_kernel(
    const float* __restrict__ R,
    const float* __restrict__ W_dt,
    const float* __restrict__ b_dt,
    float* __restrict__ dt_out) {
  const int lane  = threadIdx.x;
  const int tileN = blockIdx.x;   // 0..319
  const int tileM = blockIdx.y;   // 0..15
  const int mn    = lane & 15;
  const int khalf = (lane >> 4) * 8;

  const float* arow = R    + (size_t)(tileM * 16 + mn) * NCAT;      // P row
  const float* wrow = W_dt + (size_t)(tileN * 16 + mn) * DT_RANK;

  v8f acc = {};
  #pragma unroll
  for (int k0 = 0; k0 < DT_RANK; k0 += 32) {
    v16h a = {}, b = {};
    cvt8(a, 0, arow + k0 + khalf);
    cvt8(a, 8, arow + k0 + khalf + 16);
    cvt8(b, 0, wrow + k0 + khalf);
    cvt8(b, 8, wrow + k0 + khalf + 16);
    acc = __builtin_amdgcn_wmma_f32_16x16x32_f16(
        false, a, false, b, (short)0, acc, false, false);
  }

  const int   dcol = tileN * 16 + mn;
  const float bias = b_dt[dcol];
  #pragma unroll
  for (int v = 0; v < 8; ++v) {
    int   m   = v + ((lane >> 4) << 3);
    float val = acc[v] + bias;
    float sp  = (val > 20.f) ? val : log1pf(__expf(val));   // softplus
    dt_out[(size_t)(tileM * 16 + m) * D_INNER + dcol] = sp;
  }
}

// ---------------------------------------------------------------------------
// Kernel 3: streaming state update + reduce over n (memory-bound, ~100 MB):
//   y[b,d] = sum_n ( exp(dt*Aneg[d,n]) * h[b,d,n] + dt*x*Bm[b,n] ) * Cm[b,n]
//          + D[d]*x[b,d]
// One thread per (b,d); h/Aneg via 128-bit loads; dt read from & y written to
// the same buffer (d_out), read-before-write per thread.
// ---------------------------------------------------------------------------
__global__ __launch_bounds__(256) void ssm_pointwise_kernel(
    const float* __restrict__ x,
    const float* __restrict__ h,
    const float* __restrict__ Aneg,
    const float* __restrict__ R,
    const float* __restrict__ Dvec,
    float* __restrict__ y_dt) {
  int idx = blockIdx.x * blockDim.x + threadIdx.x;
  if (idx >= B_SZ * D_INNER) return;
  const int b = idx / D_INNER;
  const int d = idx - b * D_INNER;

  const float dtv = y_dt[idx];      // dt (written by kernel 2)
  const float xv  = x[idx];
  const float dtx = dtv * xv;

  const float4* hp = (const float4*)(h    + (size_t)idx * D_STATE);
  const float4* ap = (const float4*)(Aneg + (size_t)d   * D_STATE);
  const float4* bm = (const float4*)(R + (size_t)b * NCAT + DT_RANK);
  const float4* cm = (const float4*)(R + (size_t)b * NCAT + DT_RANK + D_STATE);

  float acc = 0.f;
  #pragma unroll
  for (int i = 0; i < 4; ++i) {
    float4 hv = hp[i], av = ap[i], bv = bm[i], cv = cm[i];
    acc += (__expf(dtv * av.x) * hv.x + dtx * bv.x) * cv.x;
    acc += (__expf(dtv * av.y) * hv.y + dtx * bv.y) * cv.y;
    acc += (__expf(dtv * av.z) * hv.z + dtx * bv.z) * cv.z;
    acc += (__expf(dtv * av.w) * hv.w + dtx * bv.w) * cv.w;
  }
  y_dt[idx] = acc + Dvec[d] * xv;
}

// ---------------------------------------------------------------------------
extern "C" void kernel_launch(void* const* d_in, const int* in_sizes, int n_in,
                              void* d_out, int out_size, void* d_ws, size_t ws_size,
                              hipStream_t stream) {
  (void)in_sizes; (void)n_in; (void)out_size; (void)ws_size;
  const float* x      = (const float*)d_in[0];  // [256,5120]
  const float* h      = (const float*)d_in[1];  // [256,5120,16]
  const float* W_x_dt = (const float*)d_in[2];  // [160,5120]
  const float* W_dt   = (const float*)d_in[3];  // [5120,160]
  const float* b_dt   = (const float*)d_in[4];  // [5120]
  const float* W_B    = (const float*)d_in[5];  // [16,5120]
  const float* W_C    = (const float*)d_in[6];  // [16,5120]
  const float* A_log  = (const float*)d_in[7];  // [5120,16]
  const float* Dvec   = (const float*)d_in[8];  // [5120]
  float*       y      = (float*)d_out;          // [256,5120] (dt, then y)

  // Workspace layout (floats): R[256*192] | Aneg[5120*16]   (~0.52 MB)
  float* R    = (float*)d_ws;
  float* Aneg = R + (size_t)B_SZ * NCAT;

  // 0) Aneg = -exp(A_log)
  {
    int n = D_INNER * D_STATE;
    aneg_kernel<<<(n + 255) / 256, 256, 0, stream>>>(A_log, Aneg, n);
  }
  // 1) R = x @ [W_x_dt | W_B | W_C]^T   (WMMA, 16x16 tiles, 4-wave K split)
  xproj_wmma_kernel<<<dim3(NCAT / 16, B_SZ / 16), 128, 0, stream>>>(
      x, W_x_dt, W_B, W_C, R);
  // 2) dt = softplus(P @ W_dt^T + b_dt) -> stored in d_out (WMMA, K=160)
  dtproj_wmma_kernel<<<dim3(D_INNER / 16, B_SZ / 16), 32, 0, stream>>>(
      R, W_dt, b_dt, y);
  // 3) streaming SSM update + output projection over n -> y in d_out
  ssm_pointwise_kernel<<<(B_SZ * D_INNER + 255) / 256, 256, 0, stream>>>(
      x, h, Aneg, R, Dvec, y);
}
